// CrossAttention_68015102100204
// MI455X (gfx1250) — matrix operand order repaired and checked by reading, not compiled, not run on hardware
//
#include <hip/hip_runtime.h>

typedef __attribute__((ext_vector_type(16))) _Float16 v16h;
typedef __attribute__((ext_vector_type(8)))  _Float16 v8h;
typedef __attribute__((ext_vector_type(4)))  _Float16 v4h;
typedef __attribute__((ext_vector_type(8)))  float    v8f;

// D = A(16x32 f16) * B(32x16 f16) + C(16x16 f32)
__device__ __forceinline__ v8f wmma16(v16h a, v16h b, v8f c) {
  return __builtin_amdgcn_wmma_f32_16x16x32_f16(false, a, false, b, (short)0, c,
                                                false, false);
}

__device__ __forceinline__ v16h gather4(const _Float16* p0, const _Float16* p1,
                                        const _Float16* p2, const _Float16* p3) {
  v4h c0 = *(const v4h*)p0;
  v4h c1 = *(const v4h*)p1;
  v4h c2 = *(const v4h*)p2;
  v4h c3 = *(const v4h*)p3;
  v16h r;
  r[0] = c0[0];  r[1] = c0[1];  r[2] = c0[2];  r[3] = c0[3];
  r[4] = c1[0];  r[5] = c1[1];  r[6] = c1[2];  r[7] = c1[3];
  r[8] = c2[0];  r[9] = c2[1];  r[10] = c2[2]; r[11] = c2[3];
  r[12] = c3[0]; r[13] = c3[1]; r[14] = c3[2]; r[15] = c3[3];
  return r;
}

// 16-bit A matrix 16x32 (ISA 7.12.2): lane&15 = row M; lanes 0-15 hold
// K {0..7, 16..23}; lanes 16-31 hold K {8..15, 24..31}.
__device__ __forceinline__ v16h loadAfrag(const _Float16* base, int stride, int kofs) {
  const int lane = threadIdx.x & 31;
  const _Float16* p = base + (lane & 15) * stride + kofs + ((lane >> 4) << 3);
  return gather4(p, p + 4, p + 16, p + 20);
}

// 16-bit B matrix 32x16: lane&15 = col N; lanes 0-15 hold
// K {0..7, 16..23}; lanes 16-31 hold K {8..15, 24..31}.
__device__ __forceinline__ v16h loadBfrag(const _Float16* base, int stride) {
  const int lane = threadIdx.x & 31;
  const _Float16* p = base + (lane & 15) * stride + ((lane >> 4) << 3);
  return gather4(p, p + 4, p + 16, p + 20);
}

__global__ __launch_bounds__(256) void convert_scale(const float* __restrict__ in,
                                                     _Float16* __restrict__ out,
                                                     int n4, float scale) {
  int i = blockIdx.x * 256 + threadIdx.x;
  if (i >= n4) return;
  float4 f = ((const float4*)in)[i];
  v4h h;
  h[0] = (_Float16)(f.x * scale);
  h[1] = (_Float16)(f.y * scale);
  h[2] = (_Float16)(f.z * scale);
  h[3] = (_Float16)(f.w * scale);
  *(v4h*)(out + (size_t)i * 4) = h;
}

// C[M x Nc] = A_f16[M x K] * W_f32[K x Nc] + bias.  64x64 tile, 128 threads,
// 4 waves in 2x2, each wave owns a 32x32 sub-tile (2x2 WMMA accumulators).
template <int OUTF32>
__global__ __launch_bounds__(128) void gemm_bias(const _Float16* __restrict__ A,
                                                 const float* __restrict__ W,
                                                 const float* __restrict__ bias,
                                                 void* __restrict__ outv,
                                                 int K, int Nc) {
  __shared__ _Float16 WsT[64][40];  // [n][k] transposed, padded
  const int tid = threadIdx.x;
  const int lane = tid & 31, wid = tid >> 5;
  const int wm = wid & 1, wn = wid >> 1;
  const int tileN = blockIdx.x * 64;
  const int tileM = blockIdx.y * 64;
  v8f acc[2][2] = {};
  // staging: thread owns column n = tid&63, k-range (tid>>6)*16 .. +15
  const int nIdx = tid & 63;
  const int kg = (tid >> 6) << 4;  // 0 or 16
  for (int kt = 0; kt < K; kt += 32) {
    float f[16];
    const float* wp = W + (size_t)(kt + kg) * Nc + tileN + nIdx;
#pragma unroll
    for (int i = 0; i < 16; ++i) f[i] = wp[(size_t)i * Nc];  // coalesced per k
    __syncthreads();
#pragma unroll
    for (int j = 0; j < 4; ++j) {
      v4h h;
      h[0] = (_Float16)f[4 * j + 0];
      h[1] = (_Float16)f[4 * j + 1];
      h[2] = (_Float16)f[4 * j + 2];
      h[3] = (_Float16)f[4 * j + 3];
      *(v4h*)&WsT[nIdx][kg + 4 * j] = h;
    }
    __syncthreads();
    const _Float16* arow0 = A + (size_t)(tileM + wm * 32) * K;
    v16h a0 = loadAfrag(arow0, K, kt);
    v16h a1 = loadAfrag(arow0 + (size_t)16 * K, K, kt);
    v16h b0 = loadBfrag(&WsT[wn * 32][0], 40);
    v16h b1 = loadBfrag(&WsT[wn * 32 + 16][0], 40);
    acc[0][0] = wmma16(a0, b0, acc[0][0]);
    acc[0][1] = wmma16(a0, b1, acc[0][1]);
    acc[1][0] = wmma16(a1, b0, acc[1][0]);
    acc[1][1] = wmma16(a1, b1, acc[1][1]);
  }
  const int hi = lane >> 4, ln = lane & 15;
  for (int ms = 0; ms < 2; ++ms)
    for (int ns = 0; ns < 2; ++ns) {
      int col = tileN + wn * 32 + ns * 16 + ln;
      float bv = bias[col];
      for (int r = 0; r < 8; ++r) {
        int row = tileM + wm * 32 + ms * 16 + r + (hi << 3);
        float v = acc[ms][ns][r] + bv;
        if (OUTF32)
          ((float*)outv)[(size_t)row * Nc + col] = v;
        else
          ((_Float16*)outv)[(size_t)row * Nc + col] = (_Float16)v;
      }
    }
}

// vT[b][h][d][n] <- kv[b*N+n][768 + h*64 + d]
__global__ __launch_bounds__(256) void transpose_v(const _Float16* __restrict__ kv,
                                                   _Float16* __restrict__ vT) {
  __shared__ _Float16 t[64][65];
  const int tid = threadIdx.x;
  const int nt = blockIdx.x & 15;
  const int bh = blockIdx.x >> 4;
  const int h = bh % 12, b = bh / 12;
  const int n0 = nt * 64;
  for (int i = 0; i < 16; ++i) {
    int e = tid + i * 256;
    int r = e >> 6, c = e & 63;
    t[r][c] = kv[(size_t)(b * 1024 + n0 + r) * 1536 + 768 + h * 64 + c];
  }
  __syncthreads();
  for (int i = 0; i < 16; ++i) {
    int e = tid + i * 256;
    int d = e >> 6, n = e & 63;
    vT[(size_t)(bh * 64 + d) * 1024 + n0 + n] = t[n][d];
  }
}

// Flash attention: grid = B*H*(N/64), 128 threads = 4 waves, wave owns 16 q-rows.
// Computes S^T = K Q^T so that softmax reductions are lane-local (one xor-16
// shuffle), and the P^T -> LDS bounce is 4x ds_store_b128 per lane.
// q already scaled by 1/sqrt(HD).
__global__ __launch_bounds__(128) void attn_kernel(const _Float16* __restrict__ qf,
                                                   const _Float16* __restrict__ kvf,
                                                   const _Float16* __restrict__ vT,
                                                   _Float16* __restrict__ wa) {
  __shared__ _Float16 Ps[4][16][72];  // per-wave P tile [q][k], padded
  __shared__ float sBuf[4][16];       // per-wave per-q scalar broadcast
  const int tid = threadIdx.x, lane = tid & 31, wid = tid >> 5;
  const int qt = blockIdx.x & 15;
  const int bh = blockIdx.x >> 4;
  const int h = bh % 12, b = bh / 12;
  const int qrow0 = qt * 64 + wid * 16;
  const int hi = lane >> 4, ln = lane & 15;

  // B operand = Q^T (contraction d contiguous per q-column); loaded once.
  const _Float16* qbase = qf + (size_t)(b * 1024 + qrow0) * 768 + h * 64;
  v16h qb0 = loadBfrag(qbase, 768);       // d 0..31
  v16h qb1 = loadBfrag(qbase + 32, 768);  // d 32..63

  float mrow = -1e30f, lrow = 0.f;  // for query q = ln (replicated in halves)
  v8f acc[4] = {};

  for (int kb = 0; kb < 16; ++kb) {
    const int k0 = kb * 64;
    // ---- S^T = K Q^T : frag ct covers keys ct*16 + (r + 8*hi), col q = ln ----
    v8f st[4];
#pragma unroll
    for (int ct = 0; ct < 4; ++ct) {
      const _Float16* kbase =
          kvf + (size_t)(b * 1024 + k0 + ct * 16) * 1536 + h * 64;
      v16h ka0 = loadAfrag(kbase, 1536, 0);   // keys x d(0..31)
      v16h ka1 = loadAfrag(kbase, 1536, 32);  // keys x d(32..63)
      v8f z = {};
      z = wmma16(ka0, qb0, z);
      z = wmma16(ka1, qb1, z);
      st[ct] = z;
    }
    // ---- softmax for column q = ln: lane-local + one xor-16 shuffle ----
    float mx = st[0][0];
#pragma unroll
    for (int ct = 0; ct < 4; ++ct)
#pragma unroll
      for (int r = 0; r < 8; ++r) mx = fmaxf(mx, st[ct][r]);
    mx = fmaxf(mx, __shfl_xor(mx, 16, 32));
    const float mn = fmaxf(mrow, mx);
    const float alpha = __expf(mrow - mn);
    mrow = mn;
    float rs = 0.f;
#pragma unroll
    for (int ct = 0; ct < 4; ++ct)
#pragma unroll
      for (int r = 0; r < 8; ++r) {
        float p = __expf(st[ct][r] - mn);
        st[ct][r] = p;
        rs += p;
      }
    rs += __shfl_xor(rs, 16, 32);
    lrow = lrow * alpha + rs;
    // ---- P^T -> LDS: 8 contiguous halves per frag = ds_store_b128 ----
#pragma unroll
    for (int ct = 0; ct < 4; ++ct) {
      v8h ph;
#pragma unroll
      for (int r = 0; r < 8; ++r) ph[r] = (_Float16)st[ct][r];
      *(v8h*)&Ps[wid][ln][ct * 16 + (hi << 3)] = ph;
    }
    sBuf[wid][ln] = alpha;  // both halves write the same value
    asm volatile("s_wait_dscnt 0" ::: "memory");
    // alpha rebroadcast to O-accumulator row layout (row = r + 8*hi)
    float4 a0 = *(const float4*)&sBuf[wid][hi << 3];
    float4 a1 = *(const float4*)&sBuf[wid][(hi << 3) + 4];
    const float ar[8] = {a0.x, a0.y, a0.z, a0.w, a1.x, a1.y, a1.z, a1.w};
#pragma unroll
    for (int ct = 0; ct < 4; ++ct)
#pragma unroll
      for (int r = 0; r < 8; ++r) acc[ct][r] *= ar[r];
    v16h pa0 = loadAfrag(&Ps[wid][0][0], 72, 0);   // P: q rows x keys 0..31
    v16h pa1 = loadAfrag(&Ps[wid][0][0], 72, 32);  // keys 32..63
    // ---- O += P V (contraction over 64 keys) ----
#pragma unroll
    for (int ct = 0; ct < 4; ++ct) {
      const _Float16* vbase = vT + ((size_t)bh * 64 + ct * 16) * 1024 + k0;
      v16h vf0 = loadBfrag(vbase, 1024);       // k 0..31
      v16h vf1 = loadBfrag(vbase + 32, 1024);  // k 32..63
      acc[ct] = wmma16(pa0, vf0, acc[ct]);
      acc[ct] = wmma16(pa1, vf1, acc[ct]);
    }
  }
  // 1/l per query (one rcp per lane), rebroadcast to row layout
  sBuf[wid][ln] = 1.0f / lrow;
  asm volatile("s_wait_dscnt 0" ::: "memory");
  float4 l0 = *(const float4*)&sBuf[wid][hi << 3];
  float4 l1 = *(const float4*)&sBuf[wid][(hi << 3) + 4];
  const float lr[8] = {l0.x, l0.y, l0.z, l0.w, l1.x, l1.y, l1.z, l1.w};
#pragma unroll
  for (int ct = 0; ct < 4; ++ct)
#pragma unroll
    for (int r = 0; r < 8; ++r) {
      float o = acc[ct][r] * lr[r];
      wa[(size_t)(b * 1024 + qrow0 + r + (hi << 3)) * 768 + h * 64 + ct * 16 + ln] =
          (_Float16)o;
    }
}

extern "C" void kernel_launch(void* const* d_in, const int* in_sizes, int n_in,
                              void* d_out, int out_size, void* d_ws, size_t ws_size,
                              hipStream_t stream) {
  const float* en = (const float*)d_in[0];
  const float* dec = (const float*)d_in[1];
  const float* Wkv = (const float*)d_in[2];
  const float* bkv = (const float*)d_in[3];
  const float* Wproj = (const float*)d_in[4];
  const float* bproj = (const float*)d_in[5];
  float* out = (float*)d_out;

  const size_t MN = 8192;  // B*N
  _Float16* enH = (_Float16*)d_ws;          // [8192][768]
  _Float16* qH = enH + MN * 768;            // [8192][768], pre-scaled by 1/8
  _Float16* kvH = qH + MN * 768;            // [8192][1536] (K | V)
  _Float16* vTH = kvH + MN * 1536;          // [B*H][64][1024]
  _Float16* waH = vTH + (size_t)96 * 64 * 1024;  // [8192][768]

  const int n4 = (int)(MN * 768 / 4);  // 1572864
  convert_scale<<<n4 / 256, 256, 0, stream>>>(en, enH, n4, 1.0f);
  convert_scale<<<n4 / 256, 256, 0, stream>>>(dec, qH, n4, 0.125f);
  gemm_bias<0><<<dim3(1536 / 64, 8192 / 64), 128, 0, stream>>>(
      enH, Wkv, bkv, (void*)kvH, 768, 1536);
  transpose_v<<<96 * 16, 256, 0, stream>>>(kvH, vTH);
  attn_kernel<<<96 * 16, 128, 0, stream>>>(qH, kvH, vTH, waH);
  gemm_bias<1><<<dim3(768 / 64, 8192 / 64), 128, 0, stream>>>(
      waH, Wproj, bproj, (void*)out, 768, 768);
}
